// RNNAgent_26121991094490
// MI455X (gfx1250) — matrix operand order repaired and checked by reading, not compiled, not run on hardware
//
#include <hip/hip_runtime.h>

typedef __attribute__((ext_vector_type(16))) _Float16 v16h;
typedef __attribute__((ext_vector_type(8)))  float    v8f;

#define N_AGENTS 32
#define IN_DIM   256
#define HID      256
#define N_ACT    64
#define BATCH    4096

#define ROWS   128          // rows per workgroup (one 16-row M-tile per wave)
#define CHUNK  64           // weight/N columns staged per panel
#define LDA    264          // f16 stride of 256-wide activation tiles (pad 8)
#define LDW    264          // f16 stride (K dir) of transposed weight panel (sync path)
#define LDSS   72           // f16 stride of 64-wide gate scratch (pad 8)

#define SZ_ACT   (ROWS * LDA * 2)          // 67584 B
#define OFF_XH   0
#define OFF_HT   (OFF_XH + SZ_ACT)
#define OFF_HNEW (OFF_HT + SZ_ACT)         // aliases X tile (phase A only)
#define OFF_X    OFF_HNEW
#define OFF_W    (OFF_HNEW + SZ_ACT)
#define SZ_W     (CHUNK * LDW * 2)         // 33792 B (async path uses 32768 of it)
#define OFF_SCR  (OFF_W + SZ_W)
#define SZ_SCR   (ROWS * LDSS * 2)         // 18432 B
#define SMEM_BYTES (OFF_SCR + 3 * SZ_SCR)  // 291840 B < 320KB

// ---- f16 transposed-weight workspace layout (element offsets) ----
#define WS_W1T_AG    (IN_DIM * HID)                                   // 65536
#define WS_G_AG      (HID * 3 * HID)                                  // 196608
#define WS_W2T_AG    (HID * N_ACT)                                    // 16384
#define WS_W1T_OFF   ((size_t)0)
#define WS_WIHT_OFF  (WS_W1T_OFF + (size_t)N_AGENTS * WS_W1T_AG)      // 2097152
#define WS_WHHT_OFF  (WS_WIHT_OFF + (size_t)N_AGENTS * WS_G_AG)      // 8388608
#define WS_W2T_OFF   (WS_WHHT_OFF + (size_t)N_AGENTS * WS_G_AG)      // 14680064
#define WS_F16_ELEMS (WS_W2T_OFF + (size_t)N_AGENTS * WS_W2T_AG)     // 15204352
#define WS_BYTES_NEEDED (WS_F16_ELEMS * 2)                            // ~29 MB

#define GLOBAL_AS __attribute__((address_space(1)))
#define LDS_AS    __attribute__((address_space(3)))

typedef int vi4 __attribute__((vector_size(16)));
typedef GLOBAL_AS vi4* async_gptr;
typedef LDS_AS    vi4* async_lptr;

__device__ __forceinline__ float sigmoidf_(float x) {
    return 1.0f / (1.0f + __expf(-x));
}

// one 16B async chunk, global -> LDS (ASYNCcnt)
__device__ __forceinline__ void async_copy_b128(const void* g, void* l) {
    __builtin_amdgcn_global_load_async_to_lds_b128(
        (async_gptr)(unsigned long long)g,
        (async_lptr)(unsigned)(unsigned long long)l, 0, 0);
}

// ---------------- sync (fallback) weight staging: f32 global -> f16 LDS, transposed ----
__device__ __forceinline__ void stage_weight(const float* __restrict__ W, int wcols,
                                             int col0, _Float16* ws, int tid) {
    int n  = tid & 63;
    int k0 = (tid >> 6) * 64;
    const float* src = W + col0 + n;
    _Float16*    dst = ws + n * LDW;
#pragma unroll 8
    for (int k = k0; k < k0 + 64; ++k)
        dst[k] = (_Float16)src[(size_t)k * wcols];
}

// ---------------- async weight staging: pre-transposed f16 panel (contiguous 32KB) ----
// LDS layout: row n (0..63) at n*256 halves; 16B unit u (0..31) stored at (u ^ (n&31)).
__device__ __forceinline__ void stage_panel_async(const _Float16* gpanel, _Float16* WS,
                                                  int tid) {
#pragma unroll
    for (int i = 0; i < 8; ++i) {
        const int c = tid + i * 256;      // 16B chunk id, 2048 total
        const int n = c >> 5;
        const int u = c & 31;
        async_copy_b128(gpanel + (size_t)c * 8,
                        WS + n * 256 + ((u ^ (n & 31)) << 3));
    }
}

// ---------------- activation staging: f32 global (stride-32 rows) -> f16 LDS ----------
__device__ __forceinline__ void stage_act(const float* __restrict__ g, long gbase,
                                          _Float16* dst, int tid) {
    int c4 = tid & 63;
    int r0 = tid >> 6;
#pragma unroll 4
    for (int rr = 0; rr < 32; ++rr) {
        int row = rr * 4 + r0;
        float4 v = *(const float4*)(g + (size_t)(gbase + 32l * row) * 256 + c4 * 4);
        _Float16* d = dst + row * LDA + c4 * 4;
        d[0] = (_Float16)v.x; d[1] = (_Float16)v.y;
        d[2] = (_Float16)v.z; d[3] = (_Float16)v.w;
    }
}

// ---------------- 16x64 += A(16x256) @ B(256x64), padded-stride B panel --------------
__device__ __forceinline__ void gemm_acc(const _Float16* act, const _Float16* ws,
                                         int mt, int lane, v8f acc[4]) {
    const int row = mt * 16 + (lane & 15);
    const int ak  = (lane >> 4) * 8;
    const int bn  = lane & 15;
    const int bk  = (lane >> 4) * 8;
    const _Float16* arow = act + row * LDA;
#pragma unroll
    for (int ks = 0; ks < 8; ++ks) {
        const int k0 = ks * 32;
        union { uint4 u[2]; v16h h; } a;
        a.u[0] = *(const uint4*)(arow + k0 + ak);
        a.u[1] = *(const uint4*)(arow + k0 + 16 + ak);
#pragma unroll
        for (int nt = 0; nt < 4; ++nt) {
            union { uint4 u[2]; v16h h; } b;
            const _Float16* bp = ws + (nt * 16 + bn) * LDW + k0 + bk;
            b.u[0] = *(const uint4*)(bp);
            b.u[1] = *(const uint4*)(bp + 16);
            acc[nt] = __builtin_amdgcn_wmma_f32_16x16x32_f16(
                false, a.h, false, b.h, (short)0, acc[nt], false, false);
        }
    }
}

// ---------------- same, but B panel is the XOR-swizzled async tile -------------------
__device__ __forceinline__ void gemm_acc_swz(const _Float16* act, const _Float16* WS,
                                             int mt, int lane, v8f acc[4]) {
    const int row = mt * 16 + (lane & 15);
    const int ak  = (lane >> 4) * 8;
    const int bn  = lane & 15;
    const int bk  = (lane >> 4) * 8;
    const _Float16* arow = act + row * LDA;
#pragma unroll
    for (int ks = 0; ks < 8; ++ks) {
        const int k0 = ks * 32;
        union { uint4 u[2]; v16h h; } a;
        a.u[0] = *(const uint4*)(arow + k0 + ak);
        a.u[1] = *(const uint4*)(arow + k0 + 16 + ak);
        const int u = (k0 + bk) >> 3;          // 16B unit within row
#pragma unroll
        for (int nt = 0; nt < 4; ++nt) {
            const int n = nt * 16 + bn;
            const int s = n & 31;
            const _Float16* rowp = WS + n * 256;
            union { uint4 u4[2]; v16h h; } b;
            b.u4[0] = *(const uint4*)(rowp + ((u ^ s) << 3));
            b.u4[1] = *(const uint4*)(rowp + (((u + 2) ^ s) << 3));
            acc[nt] = __builtin_amdgcn_wmma_f32_16x16x32_f16(
                false, a.h, false, b.h, (short)0, acc[nt], false, false);
        }
    }
}

__device__ __forceinline__ void zero_acc(v8f acc[4]) {
    const v8f z = {0.f, 0.f, 0.f, 0.f, 0.f, 0.f, 0.f, 0.f};
#pragma unroll
    for (int i = 0; i < 4; ++i) acc[i] = z;
}

// barrier; stage one 256x64 panel (async f16 tile or sync f32->f16); barrier; GEMM.
template <bool ASYNC>
__device__ __forceinline__ void panel_gemm(const float* Wg32, int wcols, int col0,
                                           const _Float16* WgT, const _Float16* act,
                                           _Float16* WS, int tid, int mt, int lane,
                                           v8f acc[4]) {
    __syncthreads();                    // previous panel's readers are done
    if constexpr (ASYNC) {
        stage_panel_async(WgT + (size_t)col0 * 256, WS, tid);
        __builtin_amdgcn_s_wait_asynccnt(0);
    } else {
        stage_weight(Wg32, wcols, col0, WS, tid);
    }
    __syncthreads();                    // panel ready (all waves contributed)
    if constexpr (ASYNC) gemm_acc_swz(act, WS, mt, lane, acc);
    else                 gemm_acc(act, WS, mt, lane, acc);
}

// ---------------- prepass: per-agent f32 [256 x ncols] -> f16 [ncols x 256] ----------
__global__ __launch_bounds__(256) void wtrans_kernel(const float* __restrict__ src,
                                                     _Float16* __restrict__ dst,
                                                     int ncols) {
    __shared__ _Float16 tile[64][72];
    const int nt = blockIdx.x, kt = blockIdx.y, a = blockIdx.z;
    const int tid = threadIdx.x;
    const float* s = src + (size_t)a * 256 * ncols + (size_t)kt * 64 * ncols + (size_t)nt * 64;
    _Float16*    d = dst + (size_t)a * ncols * 256 + (size_t)nt * 64 * 256 + kt * 64;
    {
        int n = tid & 63, k0 = tid >> 6;
#pragma unroll 4
        for (int kk = 0; kk < 16; ++kk) {
            int k = kk * 4 + k0;
            tile[k][n] = (_Float16)s[(size_t)k * ncols + n];
        }
    }
    __syncthreads();
    {
        int k = tid & 63, n0 = tid >> 6;
#pragma unroll 4
        for (int nn = 0; nn < 16; ++nn) {
            int n = nn * 4 + n0;
            d[(size_t)n * 256 + k] = tile[k][n];
        }
    }
}

// ---------------- fused agent-step kernel --------------------------------------------
template <bool ASYNC>
__global__ __launch_bounds__(256) void rnn_agent_fused(
    const float* __restrict__ x,   const float* __restrict__ h,
    const float* __restrict__ W1,  const float* __restrict__ b1,
    const float* __restrict__ Wih, const float* __restrict__ Whh,
    const float* __restrict__ bih, const float* __restrict__ bhh,
    const float* __restrict__ W2,  const float* __restrict__ b2,
    const _Float16* __restrict__ W1T, const _Float16* __restrict__ WihT,
    const _Float16* __restrict__ WhhT, const _Float16* __restrict__ W2T,
    float* __restrict__ qout, float* __restrict__ hout) {
    extern __shared__ char smem[];
    _Float16* XH   = (_Float16*)(smem + OFF_XH);
    _Float16* HT   = (_Float16*)(smem + OFF_HT);
    _Float16* HNEW = (_Float16*)(smem + OFF_HNEW);
    _Float16* XT   = (_Float16*)(smem + OFF_X);
    _Float16* WS   = (_Float16*)(smem + OFF_W);
    _Float16* SR   = (_Float16*)(smem + OFF_SCR);
    _Float16* SZg  = (_Float16*)(smem + OFF_SCR + SZ_SCR);
    _Float16* SH   = (_Float16*)(smem + OFF_SCR + 2 * SZ_SCR);

    const int tid  = threadIdx.x;
    const int lane = tid & 31;
    const int mt   = tid >> 5;
    const int agent    = blockIdx.x & 31;
    const int blockRow = blockIdx.x >> 5;
    const long gbase = agent + 32l * (blockRow * ROWS);

    const float* W1a  = W1  + (size_t)agent * IN_DIM * HID;
    const float* b1a  = b1  + agent * HID;
    const float* Wiha = Wih + (size_t)agent * HID * 3 * HID;
    const float* Whha = Whh + (size_t)agent * HID * 3 * HID;
    const float* biha = bih + agent * 3 * HID;
    const float* bhha = bhh + agent * 3 * HID;
    const float* W2a  = W2  + (size_t)agent * HID * N_ACT;
    const float* b2a  = b2  + agent * N_ACT;

    const _Float16* W1Ta  = ASYNC ? (W1T  + (size_t)agent * WS_W1T_AG) : nullptr;
    const _Float16* WihTa = ASYNC ? (WihT + (size_t)agent * WS_G_AG)   : nullptr;
    const _Float16* WhhTa = ASYNC ? (WhhT + (size_t)agent * WS_G_AG)   : nullptr;
    const _Float16* W2Ta  = ASYNC ? (W2T  + (size_t)agent * WS_W2T_AG) : nullptr;

    const int rbase = mt * 16 + 8 * (lane >> 4);
    const int cl    = lane & 15;

    // ---- Phase A: stage X and H; XH = relu(X @ W1 + b1) ----
    stage_act(x, gbase, XT, tid);
    stage_act(h, gbase, HT, tid);
    v8f acc[4];
    for (int n0 = 0; n0 < HID; n0 += CHUNK) {
        zero_acc(acc);
        panel_gemm<ASYNC>(W1a, HID, n0, W1Ta, XT, WS, tid, mt, lane, acc);
#pragma unroll
        for (int nt = 0; nt < 4; ++nt) {
            const float bv = b1a[n0 + nt * 16 + cl];
#pragma unroll
            for (int v = 0; v < 8; ++v) {
                float t = acc[nt][v] + bv;
                t = t > 0.f ? t : 0.f;
                XH[(rbase + v) * LDA + n0 + nt * 16 + cl] = (_Float16)t;
            }
        }
    }

    // ---- Phase B: GRU, per 64-column chunk ----
    for (int n0 = 0; n0 < HID; n0 += CHUNK) {
        // r = sigmoid(xh@Wih_r + h@Whh_r + bih_r + bhh_r)
        zero_acc(acc);
        panel_gemm<ASYNC>(Wiha, 3 * HID, 0 * HID + n0, WihTa, XH, WS, tid, mt, lane, acc);
        panel_gemm<ASYNC>(Whha, 3 * HID, 0 * HID + n0, WhhTa, HT, WS, tid, mt, lane, acc);
#pragma unroll
        for (int nt = 0; nt < 4; ++nt) {
            const int cc = n0 + nt * 16 + cl;
            const float bv = biha[cc] + bhha[cc];
#pragma unroll
            for (int v = 0; v < 8; ++v)
                SR[(rbase + v) * LDSS + nt * 16 + cl] = (_Float16)sigmoidf_(acc[nt][v] + bv);
        }
        // z
        zero_acc(acc);
        panel_gemm<ASYNC>(Wiha, 3 * HID, 1 * HID + n0, WihTa, XH, WS, tid, mt, lane, acc);
        panel_gemm<ASYNC>(Whha, 3 * HID, 1 * HID + n0, WhhTa, HT, WS, tid, mt, lane, acc);
#pragma unroll
        for (int nt = 0; nt < 4; ++nt) {
            const int cc = HID + n0 + nt * 16 + cl;
            const float bv = biha[cc] + bhha[cc];
#pragma unroll
            for (int v = 0; v < 8; ++v)
                SZg[(rbase + v) * LDSS + nt * 16 + cl] = (_Float16)sigmoidf_(acc[nt][v] + bv);
        }
        // h_n = h@Whh_n + bhh_n
        zero_acc(acc);
        panel_gemm<ASYNC>(Whha, 3 * HID, 2 * HID + n0, WhhTa, HT, WS, tid, mt, lane, acc);
#pragma unroll
        for (int nt = 0; nt < 4; ++nt) {
            const float bv = bhha[2 * HID + n0 + nt * 16 + cl];
#pragma unroll
            for (int v = 0; v < 8; ++v)
                SH[(rbase + v) * LDSS + nt * 16 + cl] = (_Float16)(acc[nt][v] + bv);
        }
        // i_n = xh@Wih_n + bih_n ; combine -> h_new
        zero_acc(acc);
        panel_gemm<ASYNC>(Wiha, 3 * HID, 2 * HID + n0, WihTa, XH, WS, tid, mt, lane, acc);
#pragma unroll
        for (int nt = 0; nt < 4; ++nt) {
            const float bv = biha[2 * HID + n0 + nt * 16 + cl];
#pragma unroll
            for (int v = 0; v < 8; ++v) {
                const int rl = rbase + v;
                const int c64 = nt * 16 + cl;
                const float in_ = acc[nt][v] + bv;
                const float r  = (float)SR[rl * LDSS + c64];
                const float zg = (float)SZg[rl * LDSS + c64];
                const float hn = (float)SH[rl * LDSS + c64];
                const float nn = tanhf(in_ + r * hn);
                const float hp = (float)HT[rl * LDA + n0 + c64];
                const float hnew = (1.f - zg) * nn + zg * hp;
                HNEW[rl * LDA + n0 + c64] = (_Float16)hnew;
                hout[(gbase + 32l * rl) * HID + n0 + c64] = hnew;
            }
        }
    }

    // ---- Phase C: q = h_new @ W2 + b2 ----
    zero_acc(acc);
    panel_gemm<ASYNC>(W2a, N_ACT, 0, W2Ta, HNEW, WS, tid, mt, lane, acc);
#pragma unroll
    for (int nt = 0; nt < 4; ++nt) {
        const float bv = b2a[nt * 16 + cl];
#pragma unroll
        for (int v = 0; v < 8; ++v) {
            const int rl = rbase + v;
            qout[(gbase + 32l * rl) * N_ACT + nt * 16 + cl] = acc[nt][v] + bv;
        }
    }
}

extern "C" void kernel_launch(void* const* d_in, const int* in_sizes, int n_in,
                              void* d_out, int out_size, void* d_ws, size_t ws_size,
                              hipStream_t stream) {
    (void)in_sizes; (void)n_in; (void)out_size;
    const float* x   = (const float*)d_in[0];
    const float* h   = (const float*)d_in[1];
    const float* W1  = (const float*)d_in[2];
    const float* b1  = (const float*)d_in[3];
    const float* Wih = (const float*)d_in[4];
    const float* Whh = (const float*)d_in[5];
    const float* bih = (const float*)d_in[6];
    const float* bhh = (const float*)d_in[7];
    const float* W2  = (const float*)d_in[8];
    const float* b2  = (const float*)d_in[9];

    float* qout = (float*)d_out;                                    // [131072 x 64]
    float* hout = (float*)d_out + (size_t)BATCH * N_AGENTS * N_ACT; // [131072 x 256]

    (void)hipFuncSetAttribute((const void*)rnn_agent_fused<true>,
                              hipFuncAttributeMaxDynamicSharedMemorySize, SMEM_BYTES);
    (void)hipFuncSetAttribute((const void*)rnn_agent_fused<false>,
                              hipFuncAttributeMaxDynamicSharedMemorySize, SMEM_BYTES);

    const int blocks = N_AGENTS * (BATCH / ROWS);   // 1024

    if (ws_size >= (size_t)WS_BYTES_NEEDED) {
        _Float16* wsp = (_Float16*)d_ws;
        _Float16* W1T  = wsp + WS_W1T_OFF;
        _Float16* WihT = wsp + WS_WIHT_OFF;
        _Float16* WhhT = wsp + WS_WHHT_OFF;
        _Float16* W2T  = wsp + WS_W2T_OFF;
        // prepass: transpose + convert weights once
        wtrans_kernel<<<dim3(4, 4, N_AGENTS), 256, 0, stream>>>(W1,  W1T,  256);
        wtrans_kernel<<<dim3(12, 4, N_AGENTS), 256, 0, stream>>>(Wih, WihT, 768);
        wtrans_kernel<<<dim3(12, 4, N_AGENTS), 256, 0, stream>>>(Whh, WhhT, 768);
        wtrans_kernel<<<dim3(1, 4, N_AGENTS), 256, 0, stream>>>(W2,  W2T,  64);
        rnn_agent_fused<true><<<dim3(blocks), dim3(256), SMEM_BYTES, stream>>>(
            x, h, W1, b1, Wih, Whh, bih, bhh, W2, b2,
            W1T, WihT, WhhT, W2T, qout, hout);
    } else {
        rnn_agent_fused<false><<<dim3(blocks), dim3(256), SMEM_BYTES, stream>>>(
            x, h, W1, b1, Wih, Whh, bih, bhh, W2, b2,
            nullptr, nullptr, nullptr, nullptr, qout, hout);
    }
}
